// MultiHeadSelfAttention_24172075942372
// MI455X (gfx1250) — compile-verified, matrix operand order checked by reading, not run on hardware
//
#include <hip/hip_runtime.h>
#include <hip/hip_bf16.h>

// ---------------------------------------------------------------------------
// MHA forward for MI455X (gfx1250, wave32, WMMA f32_16x16x32_f16)
// B=2, N=2048, D=1024, H=16, d=64
// Register-blocked: 32x32 per wave (GEMMs), 32 queries per wave (attention)
// ---------------------------------------------------------------------------

typedef _Float16 half8  __attribute__((ext_vector_type(8)));
typedef _Float16 half16 __attribute__((ext_vector_type(16)));
typedef float    floatx8 __attribute__((ext_vector_type(8)));

#define BATCH   2
#define SEQ     2048
#define DMODEL  1024
#define NHEADS  16
#define DHEAD   64
#define MTOT    (BATCH * SEQ)      // 4096 rows
#define QKVCOLS (3 * DMODEL)       // 3072

// Fragment load for the 16-bit A-operand layout (and B-operand when the
// source stores B's columns contiguously, i.e. transposed weight / Q rows):
//   lane L: row = L&15, g = L>>4 ;  K chunks [k0+g*8 .. +7] and [k0+16+g*8 .. +7]
__device__ __forceinline__ half16
load_frag(const _Float16* __restrict__ base, int row0, int ld, int k0, int lane) {
    int r = row0 + (lane & 15);
    int g = (lane >> 4);
    const _Float16* p = base + (size_t)r * ld + k0 + g * 8;
    half8 lo = *(const half8*)(p);
    half8 hi = *(const half8*)(p + 16);
    return __builtin_shufflevector(lo, hi, 0,1,2,3,4,5,6,7,8,9,10,11,12,13,14,15);
}

__device__ __forceinline__ floatx8
wmma_f16(half16 a, half16 b, floatx8 c) {
    // 8 args: (neg_a, A, neg_b, B, c_mod, C, reuse_a, reuse_b)
    return __builtin_amdgcn_wmma_f32_16x16x32_f16(
        false, a, false, b, (short)0, c, false, false);
}

// ---------------------------------------------------------------------------
// Prep: fp32 -> f16 elementwise
__global__ void cvt_f16_kernel(const float* __restrict__ in,
                               _Float16* __restrict__ out, int n) {
    int i = blockIdx.x * blockDim.x + threadIdx.x;
    if (i < n) out[i] = (_Float16)in[i];
}

// Prep: fp32 [rows, cols] -> f16 [cols, rows] (transpose so B columns are rows)
__global__ void transpose_f16_kernel(const float* __restrict__ in,
                                     _Float16* __restrict__ out,
                                     int rows, int cols) {
    int i = blockIdx.x * blockDim.x + threadIdx.x;
    if (i < rows * cols) {
        int r = i / cols, c = i - r * cols;
        out[(size_t)c * rows + r] = (_Float16)in[i];
    }
}

// ---------------------------------------------------------------------------
// QKV GEMM: [4096,1024]f16 x [1024,3072] -> scatter into
//   Qh, Kh : [B,H,N,d]  f16 (row-major over n,d)
//   Vt     : [B,H,d,N]  f16 (transposed: per-dcol rows contiguous over keys)
// One wave per 32x32 output tile (2x2 WMMA tiles, 2 loads/WMMA).
__global__ void qkv_gemm_kernel(const _Float16* __restrict__ xh,
                                const _Float16* __restrict__ wqkvT,
                                _Float16* __restrict__ Qh,
                                _Float16* __restrict__ Kh,
                                _Float16* __restrict__ Vt) {
    int wave = blockIdx.x * (blockDim.x >> 5) + (threadIdx.x >> 5);
    int lane = threadIdx.x & 31;
    int mt = wave & 127;          // 4096/32 = 128 row tiles
    int nt = wave >> 7;           // 3072/32 = 96  col tiles
    int m0 = mt * 32, n0 = nt * 32;

    floatx8 acc00 = {}, acc01 = {}, acc10 = {}, acc11 = {};
    #pragma unroll 2
    for (int kk = 0; kk < DMODEL; kk += 32) {
        half16 a0 = load_frag(xh,    m0,      DMODEL, kk, lane);
        half16 a1 = load_frag(xh,    m0 + 16, DMODEL, kk, lane);
        half16 b0 = load_frag(wqkvT, n0,      DMODEL, kk, lane);
        half16 b1 = load_frag(wqkvT, n0 + 16, DMODEL, kk, lane);
        acc00 = wmma_f16(a0, b0, acc00);
        acc01 = wmma_f16(a0, b1, acc01);
        acc10 = wmma_f16(a1, b0, acc10);
        acc11 = wmma_f16(a1, b1, acc11);
    }

    // C layout per tile: lane holds col = lane&15, rows r + 8*(lane>>4)
    int ho = (lane >> 4) * 8;
    #pragma unroll
    for (int tc = 0; tc < 2; ++tc) {
        int col   = n0 + tc * 16 + (lane & 15);
        int which = col >> 10;          // 0=Q 1=K 2=V (uniform per wave)
        int cc    = col & 1023;
        int h     = cc >> 6;
        int dc    = cc & 63;
        #pragma unroll
        for (int tr = 0; tr < 2; ++tr) {
            const floatx8& acc = (tc == 0) ? (tr == 0 ? acc00 : acc10)
                                           : (tr == 0 ? acc01 : acc11);
            #pragma unroll
            for (int r = 0; r < 8; ++r) {
                int m     = m0 + tr * 16 + r + ho;
                int b_idx = m >> 11;    // /2048
                int n_idx = m & 2047;
                _Float16 v = (_Float16)acc[r];
                size_t bh = (size_t)(b_idx * NHEADS + h);
                if (which == 0)      Qh[(bh * SEQ + n_idx) * DHEAD + dc] = v;
                else if (which == 1) Kh[(bh * SEQ + n_idx) * DHEAD + dc] = v;
                else                 Vt[(bh * DHEAD + dc) * SEQ + n_idx] = v;
            }
        }
    }
}

// ---------------------------------------------------------------------------
// Flash attention: one wave per (b,h, 32-query strip) = two 16-query tiles.
// Computes S^T tiles (A = K rows, B = Q^T); exp'd S^T in C-layout packs
// directly into the B operand of O^T = V^T * P^T (no LDS transpose).
// K/V fragment loads are shared between the two query tiles (1 load/WMMA).
__global__ void attn_kernel(const _Float16* __restrict__ Qh,
                            const _Float16* __restrict__ Kh,
                            const _Float16* __restrict__ Vt,
                            _Float16* __restrict__ attnout) {
    int wave = blockIdx.x * (blockDim.x >> 5) + (threadIdx.x >> 5);
    int lane = threadIdx.x & 31;
    int qs = wave & 63;           // 2048/32 = 64 query strips
    int bh = wave >> 6;           // 0..31
    int q0 = qs * 32;
    const _Float16* Qp = Qh + (size_t)bh * SEQ * DHEAD;
    const _Float16* Kp = Kh + (size_t)bh * SEQ * DHEAD;
    const _Float16* Vp = Vt + (size_t)bh * DHEAD * SEQ;

    // Q^T B-operand fragments; fixed for the whole key loop
    half16 qb[2][2];
    #pragma unroll
    for (int t = 0; t < 2; ++t) {
        qb[t][0] = load_frag(Qp, q0 + t * 16, DHEAD, 0,  lane);
        qb[t][1] = load_frag(Qp, q0 + t * 16, DHEAD, 32, lane);
    }

    floatx8 o[2][4];              // O^T accum per query tile: 64 dcols x 16 q
    #pragma unroll
    for (int t = 0; t < 2; ++t)
        #pragma unroll
        for (int mt = 0; mt < 4; ++mt) o[t][mt] = (floatx8){};

    float m_s[2] = {-1e30f, -1e30f};
    float l_s[2] = {0.0f, 0.0f};
    const float scale = 0.125f;   // 1/sqrt(64)

    for (int kt = 0; kt < SEQ; kt += 32) {
        // Shared K fragments for this 32-key step
        half16 ka00 = load_frag(Kp, kt,      DHEAD, 0,  lane);
        half16 ka01 = load_frag(Kp, kt,      DHEAD, 32, lane);
        half16 ka10 = load_frag(Kp, kt + 16, DHEAD, 0,  lane);
        half16 ka11 = load_frag(Kp, kt + 16, DHEAD, 32, lane);

        // Shared V^T fragments (16 dcols x 32 keys each)
        half16 va0 = load_frag(Vp, 0,  SEQ, kt, lane);
        half16 va1 = load_frag(Vp, 16, SEQ, kt, lane);
        half16 va2 = load_frag(Vp, 32, SEQ, kt, lane);
        half16 va3 = load_frag(Vp, 48, SEQ, kt, lane);

        #pragma unroll
        for (int t = 0; t < 2; ++t) {
            // S^T for keys kt..kt+15 / kt+16..kt+31 vs 16 queries of tile t
            floatx8 s0 = {}, s1 = {};
            s0 = wmma_f16(ka00, qb[t][0], s0);
            s0 = wmma_f16(ka01, qb[t][1], s0);
            s1 = wmma_f16(ka10, qb[t][0], s1);
            s1 = wmma_f16(ka11, qb[t][1], s1);

            float mloc = -1e30f;
            #pragma unroll
            for (int r = 0; r < 8; ++r) {
                s0[r] *= scale; s1[r] *= scale;
                mloc = fmaxf(mloc, fmaxf(s0[r], s1[r]));
            }
            mloc = fmaxf(mloc, __shfl_xor(mloc, 16, 32)); // combine key halves
            float m_new = fmaxf(m_s[t], mloc);
            float f = __expf(m_s[t] - m_new);             // first iter: 0

            float lsum = 0.0f;
            half16 pf;                                    // P^T as B operand
            #pragma unroll
            for (int r = 0; r < 8; ++r) {
                float p0 = __expf(s0[r] - m_new);
                float p1 = __expf(s1[r] - m_new);
                lsum += p0 + p1;
                pf[r]     = (_Float16)p0;  // keys kt+g*8..   -> B VGPRs 0-3
                pf[r + 8] = (_Float16)p1;  // keys kt+16+g*8  -> B VGPRs 4-7
            }
            lsum += __shfl_xor(lsum, 16, 32);
            l_s[t] = l_s[t] * f + lsum;
            #pragma unroll
            for (int mt = 0; mt < 4; ++mt)
                #pragma unroll
                for (int r = 0; r < 8; ++r) o[t][mt][r] *= f;

            // O^T += V^T(16 dcols x 32 keys) * P^T(32 keys x 16 queries)
            o[t][0] = wmma_f16(va0, pf, o[t][0]);
            o[t][1] = wmma_f16(va1, pf, o[t][1]);
            o[t][2] = wmma_f16(va2, pf, o[t][2]);
            o[t][3] = wmma_f16(va3, pf, o[t][3]);
            m_s[t] = m_new;
        }
    }

    int b = bh >> 4, h = bh & 15;
    int ho = (lane >> 4) * 8;
    #pragma unroll
    for (int t = 0; t < 2; ++t) {
        float inv = 1.0f / l_s[t];
        int q = q0 + t * 16 + (lane & 15);
        size_t rowbase = ((size_t)(b * SEQ + q)) * DMODEL + h * DHEAD;
        #pragma unroll
        for (int mt = 0; mt < 4; ++mt)
            #pragma unroll
            for (int r = 0; r < 8; ++r)
                attnout[rowbase + mt * 16 + r + ho] = (_Float16)(o[t][mt][r] * inv);
    }
}

// ---------------------------------------------------------------------------
// Output proj + residual: out = attn_h @ Wproj + x  (fp32 epilogue)
// One wave per 32x32 output tile.
__global__ void proj_kernel(const _Float16* __restrict__ ah,
                            const _Float16* __restrict__ wprojT,
                            const float* __restrict__ x,
                            float* __restrict__ out) {
    int wave = blockIdx.x * (blockDim.x >> 5) + (threadIdx.x >> 5);
    int lane = threadIdx.x & 31;
    int mt = wave & 127;          // 4096/32 = 128
    int nt = wave >> 7;           // 1024/32 = 32
    int m0 = mt * 32, n0 = nt * 32;

    floatx8 acc00 = {}, acc01 = {}, acc10 = {}, acc11 = {};
    #pragma unroll 2
    for (int kk = 0; kk < DMODEL; kk += 32) {
        half16 a0 = load_frag(ah,     m0,      DMODEL, kk, lane);
        half16 a1 = load_frag(ah,     m0 + 16, DMODEL, kk, lane);
        half16 b0 = load_frag(wprojT, n0,      DMODEL, kk, lane);
        half16 b1 = load_frag(wprojT, n0 + 16, DMODEL, kk, lane);
        acc00 = wmma_f16(a0, b0, acc00);
        acc01 = wmma_f16(a0, b1, acc01);
        acc10 = wmma_f16(a1, b0, acc10);
        acc11 = wmma_f16(a1, b1, acc11);
    }

    int ho = (lane >> 4) * 8;
    #pragma unroll
    for (int tc = 0; tc < 2; ++tc) {
        int col = n0 + tc * 16 + (lane & 15);
        #pragma unroll
        for (int tr = 0; tr < 2; ++tr) {
            const floatx8& acc = (tc == 0) ? (tr == 0 ? acc00 : acc10)
                                           : (tr == 0 ? acc01 : acc11);
            #pragma unroll
            for (int r = 0; r < 8; ++r) {
                size_t idx = (size_t)(m0 + tr * 16 + r + ho) * DMODEL + col;
                out[idx] = acc[r] + x[idx];
            }
        }
    }
}

// ---------------------------------------------------------------------------
extern "C" void kernel_launch(void* const* d_in, const int* in_sizes, int n_in,
                              void* d_out, int out_size, void* d_ws, size_t ws_size,
                              hipStream_t stream) {
    const float* x      = (const float*)d_in[0];  // [2,2048,1024]
    const float* w_qkv  = (const float*)d_in[1];  // [1024,3072]
    const float* w_proj = (const float*)d_in[2];  // [1024,1024]
    float* out = (float*)d_out;

    // f16 workspace layout (element counts)
    const size_t N_XH   = (size_t)MTOT * DMODEL;        // 4,194,304
    const size_t N_WQKV = (size_t)DMODEL * QKVCOLS;     // 3,145,728
    const size_t N_WPRJ = (size_t)DMODEL * DMODEL;      // 1,048,576
    const size_t N_HEADBUF = (size_t)BATCH * NHEADS * SEQ * DHEAD; // 4,194,304

    _Float16* xh     = (_Float16*)d_ws;
    _Float16* wqkvT  = xh    + N_XH;
    _Float16* wprojT = wqkvT + N_WQKV;
    _Float16* Qh     = wprojT + N_WPRJ;
    _Float16* Kh     = Qh + N_HEADBUF;
    _Float16* Vt     = Kh + N_HEADBUF;
    _Float16* ah     = Vt + N_HEADBUF;                  // [4096,1024] f16

    // Prep: casts + weight transposes
    cvt_f16_kernel<<<(int)(N_XH / 256), 256, 0, stream>>>(x, xh, (int)N_XH);
    transpose_f16_kernel<<<(int)(N_WQKV / 256), 256, 0, stream>>>(
        w_qkv, wqkvT, DMODEL, QKVCOLS);
    transpose_f16_kernel<<<(int)(N_WPRJ / 256), 256, 0, stream>>>(
        w_proj, wprojT, DMODEL, DMODEL);

    // QKV GEMM: (4096/32)*(3072/32) = 12288 waves, 8 waves/block
    qkv_gemm_kernel<<<12288 / 8, 256, 0, stream>>>(xh, wqkvT, Qh, Kh, Vt);

    // Attention: 32 (b,h) * 64 query strips = 2048 waves
    attn_kernel<<<2048 / 8, 256, 0, stream>>>(Qh, Kh, Vt, ah);

    // Proj + residual: (4096/32)*(1024/32) = 4096 waves
    proj_kernel<<<4096 / 8, 256, 0, stream>>>(ah, wprojT, x, out);
}
